// SDCDMNet_51161650430117
// MI455X (gfx1250) — matrix operand order repaired and checked by reading, not compiled, not run on hardware
//
#include <hip/hip_runtime.h>
#include <hip/hip_bf16.h>

// Problem constants (match reference)
#define BB   4096
#define KK   256
#define HH   64
// pre_b GEMM: (B x 128) f16  @  (128 x 64) f16  -> (B x 64) f32

typedef __attribute__((ext_vector_type(16))) _Float16 v16h;
typedef __attribute__((ext_vector_type(8)))  float    v8f;

__device__ __forceinline__ float sigmoidf(float x) {
    return 1.0f / (1.0f + __expf(-x));
}

// ---------------------------------------------------------------------------
// Kernel A: pack A = [v_habit | e_j] (B x 128) and Bmat = W1[:128] (128 x 64)
// into the wave32 WMMA-native f16 layouts (CDNA5 ISA §7.12.2).
// A tile layout:  idx = ((mt*4+kt)*32 + lane)*16 + i
//   lane<16:  M = mt*16+lane,     i<8 -> K=kt*32+i,      i>=8 -> K=kt*32+i+8
//   lane>=16: M = mt*16+lane-16,  i<8 -> K=kt*32+i+8,    i>=8 -> K=kt*32+i+16
// B tile layout:  idx = ((nt*4+kt)*32 + lane)*16 + i
//   N = nt*16 + lane%16,  K = kt*32 + 8*(lane/16) + i + (i>=8 ? 8 : 0)  (same K order as A)
// ---------------------------------------------------------------------------
__global__ void pack_ab_kernel(const int* __restrict__ stu_id,
                               const int* __restrict__ exer_id,
                               const float* __restrict__ W_habit,
                               const float* __restrict__ W_exer,
                               const float* __restrict__ W1,
                               _Float16* __restrict__ Apack,
                               _Float16* __restrict__ Bpack) {
    int tid = blockIdx.x * blockDim.x + threadIdx.x;

    if (tid < BB * 128) {
        int i    = tid & 15;
        int l    = (tid >> 4) & 31;
        int kt   = (tid >> 9) & 3;
        int mt   = tid >> 11;
        int m    = mt * 16 + (l & 15);
        int half = l >> 4;
        int kk   = (i < 8) ? (i + 8 * half) : (i + 8 + 8 * half);
        int K    = kt * 32 + kk;
        float v;
        if (K < HH) v = W_habit[stu_id[m] * HH + K];
        else        v = W_exer[exer_id[m] * HH + (K - HH)];
        Apack[tid] = (_Float16)v;
    }
    if (tid < 4 * 4 * 32 * 16) {   // 8192 elements of Bpack
        int i  = tid & 15;
        int l  = (tid >> 4) & 31;
        int kt = (tid >> 9) & 3;
        int nt = tid >> 11;
        int n  = nt * 16 + (l & 15);
        int k  = kt * 32 + ((i < 8) ? (i + 8 * (l >> 4)) : (i + 8 + 8 * (l >> 4)));
        Bpack[tid] = (_Float16)W1[k * HH + n];
    }
}

// ---------------------------------------------------------------------------
// Kernel B: pre_b = Apack @ Bpack + b1   via v_wmma_f32_16x16x32_f16.
// One wave per 16x16 output tile; 4 waves/block cover the 4 N-tiles.
// D layout: VGPR r, lane l -> M = r + 8*(l/16), N = l%16.
// ---------------------------------------------------------------------------
__global__ void __launch_bounds__(128)
gemm_preb_kernel(const _Float16* __restrict__ Apack,
                 const _Float16* __restrict__ Bpack,
                 const float* __restrict__ b1,
                 float* __restrict__ pre_b) {
    int mt   = blockIdx.x;           // 0..255
    int nt   = threadIdx.x >> 5;     // 0..3
    int lane = threadIdx.x & 31;

    v8f c = {};
#pragma unroll
    for (int kt = 0; kt < 4; ++kt) {
        v16h a = *(const v16h*)(Apack + (((mt * 4 + kt) * 32 + lane) << 4));
        v16h b = *(const v16h*)(Bpack + (((nt * 4 + kt) * 32 + lane) << 4));
        c = __builtin_amdgcn_wmma_f32_16x16x32_f16(
                /*neg_a=*/false, a, /*neg_b=*/false, b,
                /*c_mod=*/(short)0, c, /*reuse_a=*/false, /*reuse_b=*/false);
    }

    int n     = nt * 16 + (lane & 15);
    int mbase = mt * 16 + 8 * (lane >> 4);
    float bias = b1[n];
#pragma unroll
    for (int r = 0; r < 8; ++r)
        pre_b[(mbase + r) * HH + n] = c[r] + bias;
}

// ---------------------------------------------------------------------------
// Kernel C: pre_kT[h][k] = sum_j W_conc[k][j] * W1[(128+j)*64 + h]
// Stored transposed (H x K) so the fused kernel reads it fully coalesced.
// ---------------------------------------------------------------------------
__global__ void prek_kernel(const float* __restrict__ W_conc,
                            const float* __restrict__ W1,
                            float* __restrict__ pre_kT) {
    int tid = blockIdx.x * blockDim.x + threadIdx.x;   // 0..16383
    int k = tid & (KK - 1);
    int h = tid >> 8;
    float s = 0.0f;
#pragma unroll 8
    for (int j = 0; j < HH; ++j)
        s += W_conc[k * HH + j] * W1[(2 * HH + j) * HH + h];
    pre_kT[h * KK + k] = s;
}

// ---------------------------------------------------------------------------
// Kernel D: fused attention + dual-scoring. One block per batch row,
// one thread per knowledge point k (256 threads = 8 wave32).
// Outputs (concatenated): pred[B], h_pos[B*K], h_neg[B*K], kp[B*K] (as f32).
// ---------------------------------------------------------------------------
__global__ void __launch_bounds__(KK)
fused_kernel(const int* __restrict__ stu_id,
             const int* __restrict__ exer_id,
             const int* __restrict__ kp,
             const float* __restrict__ W_pos,
             const float* __restrict__ W_neg,
             const float* __restrict__ W_diff,
             const float* __restrict__ W_trap,
             const float* __restrict__ W2,
             const float* __restrict__ b2,
             const float* __restrict__ pre_b,
             const float* __restrict__ pre_kT,
             float* __restrict__ out) {
    __shared__ float s_preb[HH];
    __shared__ float s_w2[HH];
    __shared__ float s_red[KK];

    const int b = blockIdx.x;
    const int k = threadIdx.x;

    const int stu = stu_id[b];
    const int ex  = exer_id[b];

    if (k < HH) {
        s_preb[k] = pre_b[b * HH + k];
        s_w2[k]   = W2[k];
    }
    __syncthreads();

    // a[k] = sum_h relu(pre_b[b,h] + pre_k[k,h]) * W2[h] + b2
    float a = 0.0f;
#pragma unroll 8
    for (int h = 0; h < HH; ++h) {
        float v = s_preb[h] + pre_kT[h * KK + k];
        a = fmaf(fmaxf(v, 0.0f), s_w2[h], a);
    }
    a += b2[0];

    const int   kpv = kp[b * KK + k];
    const float ninf = -__builtin_inff();
    float am = kpv ? a : ninf;

    // block max
    s_red[k] = am;
    __syncthreads();
#pragma unroll
    for (int s = KK / 2; s > 0; s >>= 1) {
        if (k < s) s_red[k] = fmaxf(s_red[k], s_red[k + s]);
        __syncthreads();
    }
    float m = s_red[0];
    __syncthreads();

    // exp + block sum   (all-masked row -> softmax NaN -> nan_to_num -> 0)
    float e = (kpv && m != ninf) ? __expf(am - m) : 0.0f;
    s_red[k] = e;
    __syncthreads();
#pragma unroll
    for (int s = KK / 2; s > 0; s >>= 1) {
        if (k < s) s_red[k] += s_red[k + s];
        __syncthreads();
    }
    float ssum = s_red[0];
    __syncthreads();
    float alpha = (ssum > 0.0f) ? (e / ssum) : 0.0f;

    // dual mastery + outputs
    float hp = sigmoidf(W_pos[stu * KK + k]);
    float hn = sigmoidf(W_neg[stu * KK + k]);
    float mu = sigmoidf(W_trap[ex]);

    float* out_hpos = out + BB;
    float* out_hneg = out + BB + BB * KK;
    float* out_kp   = out + BB + 2 * BB * KK;
    out_hpos[b * KK + k] = hp;
    out_hneg[b * KK + k] = hn;
    out_kp[b * KK + k]   = (float)kpv;

    s_red[k] = alpha * (hp - mu * hn);
    __syncthreads();
#pragma unroll
    for (int s = KK / 2; s > 0; s >>= 1) {
        if (k < s) s_red[k] += s_red[k + s];
        __syncthreads();
    }
    if (k == 0) {
        float dj = sigmoidf(W_diff[ex]);
        out[b] = sigmoidf(s_red[0] - dj);
    }
}

// ---------------------------------------------------------------------------
extern "C" void kernel_launch(void* const* d_in, const int* in_sizes, int n_in,
                              void* d_out, int out_size, void* d_ws, size_t ws_size,
                              hipStream_t stream) {
    const int*   stu_id  = (const int*)d_in[0];
    const int*   exer_id = (const int*)d_in[1];
    const int*   kp      = (const int*)d_in[2];
    const float* W_pos   = (const float*)d_in[3];
    const float* W_neg   = (const float*)d_in[4];
    const float* W_diff  = (const float*)d_in[5];
    const float* W_trap  = (const float*)d_in[6];
    const float* W_habit = (const float*)d_in[7];
    const float* W_exer  = (const float*)d_in[8];
    const float* W_conc  = (const float*)d_in[9];
    const float* W1      = (const float*)d_in[10];
    const float* b1      = (const float*)d_in[11];
    const float* W2      = (const float*)d_in[12];
    const float* b2      = (const float*)d_in[13];
    float* out = (float*)d_out;

    // workspace layout (16B aligned throughout)
    char* ws = (char*)d_ws;
    float*    pre_b  = (float*)ws;                                      // B*H   f32 = 1 MB
    float*    pre_kT = (float*)(ws + (size_t)BB * HH * 4);              // H*K   f32 = 64 KB
    _Float16* Apack  = (_Float16*)(ws + (size_t)BB * HH * 4 + HH * KK * 4);          // B*128 f16 = 1 MB
    _Float16* Bpack  = (_Float16*)(ws + (size_t)BB * HH * 4 + HH * KK * 4
                                      + (size_t)BB * 128 * 2);          // 128*64 f16 = 16 KB

    // A: pack operands into WMMA layout
    pack_ab_kernel<<<(BB * 128 + 255) / 256, 256, 0, stream>>>(
        stu_id, exer_id, W_habit, W_exer, W1, Apack, Bpack);

    // B: pre_b = [v_habit|e_j] @ W1[:128] + b1   (v_wmma_f32_16x16x32_f16)
    gemm_preb_kernel<<<BB / 16, 128, 0, stream>>>(Apack, Bpack, b1, pre_b);

    // C: pre_kT = (W_conc @ W1[2H:])^T
    prek_kernel<<<(HH * KK + 255) / 256, 256, 0, stream>>>(W_conc, W1, pre_kT);

    // D: fused softmax-attention scoring + all outputs
    fused_kernel<<<BB, KK, 0, stream>>>(stu_id, exer_id, kp,
                                        W_pos, W_neg, W_diff, W_trap,
                                        W2, b2, pre_b, pre_kT, out);
}
